// MultiHeadAttention_40647570489512
// MI455X (gfx1250) — matrix operand order repaired and checked by reading, not compiled, not run on hardware
//
#include <hip/hip_runtime.h>
#include <hip/hip_bf16.h>
#include <math.h>

// ---------------------------------------------------------------------------
// MI455X (gfx1250) fused multi-head attention block.
// wave32, WMMA 16x16x32 bf16 (fp32 accum), async global->LDS staging.
// ---------------------------------------------------------------------------

typedef __bf16 bf16;
typedef __attribute__((ext_vector_type(16))) __bf16 v16bf;
typedef __attribute__((ext_vector_type(8)))  __bf16 v8bf;
typedef __attribute__((ext_vector_type(8)))  float  v8f;

#define D_MODEL 1024
#define NUM_HEADS 16
#define D_K 64
#define SEQ 2048
#define BATCH 2
#define ROWS (BATCH * SEQ)   // 4096
#define LN_EPS 1e-5f

// -------------------- WMMA helpers --------------------

__device__ __forceinline__ v8f wmma_bf16(v16bf a, v16bf b, v8f c) {
  return __builtin_amdgcn_wmma_f32_16x16x32_bf16(
      /*neg_a=*/false, a, /*neg_b=*/false, b,
      /*c_mod=*/(short)0, c, /*reuse_a=*/false, /*reuse_b=*/false);
}

// A fragment from bf16 source (global or LDS). p = row base + k0.
// Lane m = lane&15 (baked into p by caller); half = lane>>4 holds
// contiguous K = half*16 + [0..15].
__device__ __forceinline__ v16bf frag_a_bf16(const bf16* p, int half) {
  v8bf lo = *reinterpret_cast<const v8bf*>(p + half * 16);
  v8bf hi = *reinterpret_cast<const v8bf*>(p + half * 16 + 8);
  v16bf a;
#pragma unroll
  for (int i = 0; i < 8; ++i) { a[i] = lo[i]; a[8 + i] = hi[i]; }
  return a;
}

// B fragment from bf16 source where each memory row owns one B column.
// half holds contiguous K = half*16 + [0..15].
__device__ __forceinline__ v16bf frag_b_bf16(const bf16* p, int half) {
  return *reinterpret_cast<const v16bf*>(p + half * 16);
}

// -------------------- async global -> LDS copy --------------------
// Copies a 128-row x 32-col bf16 tile (8 KB) from global (row stride =
// D_MODEL elements) into a contiguous LDS tile, using the CDNA5 async
// copy path (tracked by ASYNCcnt). 256 threads x 2 x 16B chunks.
__device__ __forceinline__ void async_copy_tile(bf16* lds_tile,
                                                const bf16* gsrc_rows,
                                                int k0, int tid) {
  const unsigned lds_base = (unsigned)(uintptr_t)lds_tile;  // low 32b = LDS offset
#pragma unroll
  for (int j = 0; j < 2; ++j) {
    const int c = tid + j * 256;           // chunk id 0..511 (16B each)
    const int row = c >> 2, q = c & 3;
    unsigned lds_off = lds_base + (unsigned)c * 16u;
    unsigned long long gaddr =
        (unsigned long long)(uintptr_t)(gsrc_rows + (size_t)row * D_MODEL + k0 + q * 8);
    asm volatile("global_load_async_to_lds_b128 %0, %1, off"
                 :: "v"(lds_off), "v"(gaddr) : "memory");
  }
}

__device__ __forceinline__ void wait_async0() {
  asm volatile("s_wait_asynccnt 0x0" ::: "memory");
}

// ---------------------------------------------------------------------------
// Kernel 0: fp32 -> bf16 conversion (one-time; weights/activations).
// ---------------------------------------------------------------------------
__global__ void __launch_bounds__(256)
cvt_kernel(const float* __restrict__ in, bf16* __restrict__ out, int n) {
  const int i = (blockIdx.x * 256 + threadIdx.x) * 4;
  if (i < n) {
    const float4 v = *reinterpret_cast<const float4*>(in + i);
    out[i + 0] = (bf16)v.x;
    out[i + 1] = (bf16)v.y;
    out[i + 2] = (bf16)v.z;
    out[i + 3] = (bf16)v.w;
  }
}

// ---------------------------------------------------------------------------
// Kernel 1: projection GEMM  out = X @ W^T + bias (bf16 in, bf16 out).
// M=4096, N=1024, K=1024. Block = 8 waves, block tile 128x128, wave tile
// 32x64. A/B tiles double-buffered in LDS via global_load_async_to_lds.
// mode: 0/1 -> [b,h,s,dk], 2 -> V^T [b,h,dk,s]
// ---------------------------------------------------------------------------
__global__ void __launch_bounds__(256)
proj_kernel(const bf16* __restrict__ X, const bf16* __restrict__ W,
            const float* __restrict__ bias, bf16* __restrict__ out, int mode) {
  __shared__ bf16 sA[2][128 * 32];   // [row][k], row stride 32 elems (64B)
  __shared__ bf16 sB[2][128 * 32];   // [col][k]

  const int tid  = threadIdx.x;
  const int lane = tid & 31;
  const int wave = tid >> 5;
  const int half = lane >> 4;
  const int l16  = lane & 15;
  const int mblk = blockIdx.y * 128;
  const int nblk = blockIdx.x * 128;
  const int wm = (wave >> 1) * 32;   // wave row offset inside block tile
  const int wn = (wave & 1) * 64;    // wave col offset inside block tile

  const bf16* Arows = X + (size_t)mblk * D_MODEL;
  const bf16* Brows = W + (size_t)nblk * D_MODEL;

  v8f acc[2][4] = {};

  // prologue: stage k-step 0
  async_copy_tile(sA[0], Arows, 0, tid);
  async_copy_tile(sB[0], Brows, 0, tid);

  for (int ks = 0; ks < 32; ++ks) {
    const int cur = ks & 1;
    wait_async0();        // own async writes for buf[cur] done
    __syncthreads();      // everyone's writes done; everyone done reading buf[cur^1]
    if (ks + 1 < 32) {    // prefetch next k-step into the other buffer
      async_copy_tile(sA[cur ^ 1], Arows, (ks + 1) * 32, tid);
      async_copy_tile(sB[cur ^ 1], Brows, (ks + 1) * 32, tid);
    }

    const bf16* a_base = sA[cur];
    const bf16* b_base = sB[cur];
    v16bf a0 = frag_a_bf16(a_base + (wm + l16) * 32, half);
    v16bf a1 = frag_a_bf16(a_base + (wm + 16 + l16) * 32, half);
    v16bf b0 = frag_b_bf16(b_base + (wn + 0  + l16) * 32, half);
    v16bf b1 = frag_b_bf16(b_base + (wn + 16 + l16) * 32, half);
    v16bf b2 = frag_b_bf16(b_base + (wn + 32 + l16) * 32, half);
    v16bf b3 = frag_b_bf16(b_base + (wn + 48 + l16) * 32, half);
    acc[0][0] = wmma_bf16(a0, b0, acc[0][0]);
    acc[0][1] = wmma_bf16(a0, b1, acc[0][1]);
    acc[0][2] = wmma_bf16(a0, b2, acc[0][2]);
    acc[0][3] = wmma_bf16(a0, b3, acc[0][3]);
    acc[1][0] = wmma_bf16(a1, b0, acc[1][0]);
    acc[1][1] = wmma_bf16(a1, b1, acc[1][1]);
    acc[1][2] = wmma_bf16(a1, b2, acc[1][2]);
    acc[1][3] = wmma_bf16(a1, b3, acc[1][3]);
  }

#pragma unroll
  for (int i = 0; i < 2; ++i) {
#pragma unroll
    for (int j = 0; j < 4; ++j) {
      const int col = nblk + wn + j * 16 + l16;
      const float bv = bias[col];
      const int hh = col >> 6, dk = col & 63;
#pragma unroll
      for (int r = 0; r < 8; ++r) {
        const int row = mblk + wm + i * 16 + r + 8 * half;  // 0..4095
        const int bb = row >> 11, s = row & (SEQ - 1);
        const float v = acc[i][j][r] + bv;
        size_t off;
        if (mode == 2)  // V^T: [b][h][dk][s]
          off = (((size_t)(bb * NUM_HEADS + hh) * D_K + dk) * SEQ + s);
        else            // Q/K: [b][h][s][dk]
          off = (((size_t)(bb * NUM_HEADS + hh) * SEQ + s) * D_K + dk);
        out[off] = (bf16)v;
      }
    }
  }
}

// ---------------------------------------------------------------------------
// Kernel 2: fused scores + softmax + attn write + ctx = attn @ V.
// One block (8 waves) per (b, h, 16-query-row tile). Each wave owns a
// 16x256 key strip in registers (16 WMMA accumulators).
// Dynamic LDS: bf16 attn tile [16][2064] + fp32 ctx [16][64] + reduce[8][16].
// ---------------------------------------------------------------------------
#define ATTN_STRIDE 2064   // 2048 + 16 pad: rows stay 32B-aligned, dodges banks

__global__ void __launch_bounds__(256)
attn_kernel(const bf16* __restrict__ Qb, const bf16* __restrict__ Kb,
            const bf16* __restrict__ Vt, float* __restrict__ attn_out,
            bf16* __restrict__ ctx_out) {
  extern __shared__ char smem_raw[];
  bf16*  attn_lds = (bf16*)smem_raw;                                   // 66048 B
  float* ctx_lds  = (float*)(smem_raw + 16 * ATTN_STRIDE * 2);         //  4096 B
  float* red      = (float*)(smem_raw + 16 * ATTN_STRIDE * 2 + 4096);  //   512 B

  const int gid = blockIdx.x;
  const int qt = gid & 127;
  const int h  = (gid >> 7) & (NUM_HEADS - 1);
  const int b  = gid >> 11;
  const int q0 = qt * 16;

  const int lane = threadIdx.x & 31;
  const int wave = threadIdx.x >> 5;
  const int half = lane >> 4;
  const int l16  = lane & 15;
  const int kc0  = wave * 256;

  const size_t head = (size_t)(b * NUM_HEADS + h) * SEQ * D_K;
  const bf16* Qh = Qb + head;
  const bf16* Kh = Kb + head;
  const bf16* Vh = Vt + head;   // [dk][s] within head

  // --- scores: S = Q K^T / sqrt(Dk) ---
  const bf16* qrow = Qh + (size_t)(q0 + l16) * D_K;
  const v16bf aq0 = frag_a_bf16(qrow, half);       // dk 0..31
  const v16bf aq1 = frag_a_bf16(qrow + 32, half);  // dk 32..63

  v8f acc[16] = {};
#pragma unroll
  for (int t = 0; t < 16; ++t) {
    const bf16* krow = Kh + (size_t)(kc0 + t * 16 + l16) * D_K;
    v16bf b0 = frag_b_bf16(krow, half);
    v16bf b1 = frag_b_bf16(krow + 32, half);
    acc[t] = wmma_bf16(aq0, b0, acc[t]);
    acc[t] = wmma_bf16(aq1, b1, acc[t]);
  }
  const float scale = 0.125f;  // 1/sqrt(64)
#pragma unroll
  for (int t = 0; t < 16; ++t)
#pragma unroll
    for (int r = 0; r < 8; ++r) acc[t][r] = acc[t][r] * scale;

  // --- softmax: row max across 2048 keys ---
  float rmax[8];
#pragma unroll
  for (int r = 0; r < 8; ++r) {
    float m = -1e30f;
#pragma unroll
    for (int t = 0; t < 16; ++t) m = fmaxf(m, acc[t][r]);
    rmax[r] = m;
  }
#pragma unroll
  for (int msk = 1; msk < 16; msk <<= 1)
#pragma unroll
    for (int r = 0; r < 8; ++r)
      rmax[r] = fmaxf(rmax[r], __shfl_xor(rmax[r], msk, 32));
  if (l16 == 0)
#pragma unroll
    for (int r = 0; r < 8; ++r) red[wave * 16 + r + 8 * half] = rmax[r];
  __syncthreads();
  float gmax[8];
#pragma unroll
  for (int r = 0; r < 8; ++r) {
    float m = -1e30f;
#pragma unroll
    for (int wv = 0; wv < 8; ++wv) m = fmaxf(m, red[wv * 16 + r + 8 * half]);
    gmax[r] = m;
  }
  __syncthreads();

  // --- exp + row sum ---
  float rsum[8] = {0.f, 0.f, 0.f, 0.f, 0.f, 0.f, 0.f, 0.f};
#pragma unroll
  for (int t = 0; t < 16; ++t)
#pragma unroll
    for (int r = 0; r < 8; ++r) {
      float e = __expf(acc[t][r] - gmax[r]);
      acc[t][r] = e;
      rsum[r] += e;
    }
#pragma unroll
  for (int msk = 1; msk < 16; msk <<= 1)
#pragma unroll
    for (int r = 0; r < 8; ++r) rsum[r] += __shfl_xor(rsum[r], msk, 32);
  if (l16 == 0)
#pragma unroll
    for (int r = 0; r < 8; ++r) red[wave * 16 + r + 8 * half] = rsum[r];
  __syncthreads();
  float rinv[8];
#pragma unroll
  for (int r = 0; r < 8; ++r) {
    float s = 0.f;
#pragma unroll
    for (int wv = 0; wv < 8; ++wv) s += red[wv * 16 + r + 8 * half];
    rinv[r] = 1.0f / s;
  }

  // zero ctx accumulation buffer
  for (int i = threadIdx.x; i < 16 * D_K; i += 256) ctx_lds[i] = 0.f;

  // --- normalize; write attn (fp32: the one mandatory HBM write) + LDS bf16 ---
  const size_t attn_base = ((size_t)(b * NUM_HEADS + h) * SEQ + q0) * SEQ;
#pragma unroll
  for (int t = 0; t < 16; ++t) {
    const int col = kc0 + t * 16 + l16;
#pragma unroll
    for (int r = 0; r < 8; ++r) {
      const int rr = r + 8 * half;
      const float v = acc[t][r] * rinv[r];
      attn_out[attn_base + (size_t)rr * SEQ + col] = v;
      attn_lds[rr * ATTN_STRIDE + col] = (bf16)v;
    }
  }
  __syncthreads();

  // --- ctx = attn @ V : each wave covers its 256-key strip, all 64 dk cols ---
  v8f c2[4] = {};
#pragma unroll
  for (int ks = 0; ks < 8; ++ks) {
    const int k0 = kc0 + ks * 32;
    v16bf a = frag_a_bf16(attn_lds + l16 * ATTN_STRIDE + k0, half);
#pragma unroll
    for (int t2 = 0; t2 < 4; ++t2) {
      const bf16* vrow = Vh + (size_t)(t2 * 16 + l16) * SEQ + k0;  // V^T rows
      v16bf bb = frag_b_bf16(vrow, half);
      c2[t2] = wmma_bf16(a, bb, c2[t2]);
    }
  }
#pragma unroll
  for (int t2 = 0; t2 < 4; ++t2)
#pragma unroll
    for (int r = 0; r < 8; ++r)
      atomicAdd(&ctx_lds[(r + 8 * half) * D_K + t2 * 16 + l16], c2[t2][r]);
  __syncthreads();

  // --- write ctx tile bf16 to workspace: [b][s][h*64+dk] ---
  for (int i = threadIdx.x; i < 16 * D_K; i += 256) {
    const int m = i >> 6, dk = i & 63;
    ctx_out[((size_t)(b * SEQ) + q0 + m) * D_MODEL + h * D_K + dk] =
        (bf16)ctx_lds[i];
  }
}

// ---------------------------------------------------------------------------
// Kernel 3: out = LayerNorm(ctx @ Wo^T + bo + residual).
// One block per 16 rows; 8 waves each own 128 columns (8 WMMA tiles),
// so the full 1024-wide row lives in registers for the LN reduction.
// ---------------------------------------------------------------------------
__global__ void __launch_bounds__(256)
outproj_ln_kernel(const bf16* __restrict__ ctx, const bf16* __restrict__ Wo,
                  const float* __restrict__ bo, const float* __restrict__ query,
                  const float* __restrict__ gamma, const float* __restrict__ beta,
                  float* __restrict__ out) {
  __shared__ float ssum[8 * 16];
  __shared__ float ssq[8 * 16];

  const int lane = threadIdx.x & 31;
  const int wave = threadIdx.x >> 5;
  const int half = lane >> 4;
  const int l16  = lane & 15;
  const int m0 = blockIdx.x * 16;
  const int n0 = wave * 128;

  const bf16* arow = ctx + (size_t)(m0 + l16) * D_MODEL;
  v8f acc[8] = {};
  for (int k0 = 0; k0 < D_MODEL; k0 += 32) {
    __builtin_prefetch(arow + k0 + 128, 0, 0);   // global_prefetch_b8
    v16bf a = frag_a_bf16(arow + k0, half);
#pragma unroll
    for (int t = 0; t < 8; ++t) {
      const bf16* wrow = Wo + (size_t)(n0 + t * 16 + l16) * D_MODEL + k0;
      v16bf b = frag_b_bf16(wrow, half);
      acc[t] = wmma_bf16(a, b, acc[t]);
    }
  }

  // bias + residual, accumulate per-row sum / sumsq
  float psum[8] = {0.f, 0.f, 0.f, 0.f, 0.f, 0.f, 0.f, 0.f};
  float psq[8]  = {0.f, 0.f, 0.f, 0.f, 0.f, 0.f, 0.f, 0.f};
#pragma unroll
  for (int t = 0; t < 8; ++t) {
    const int col = n0 + t * 16 + l16;
    const float bv = bo[col];
#pragma unroll
    for (int r = 0; r < 8; ++r) {
      const int row = m0 + r + 8 * half;
      const float v = acc[t][r] + bv + query[(size_t)row * D_MODEL + col];
      acc[t][r] = v;
      psum[r] += v;
      psq[r]  += v * v;
    }
  }
#pragma unroll
  for (int msk = 1; msk < 16; msk <<= 1)
#pragma unroll
    for (int r = 0; r < 8; ++r) {
      psum[r] += __shfl_xor(psum[r], msk, 32);
      psq[r]  += __shfl_xor(psq[r],  msk, 32);
    }
  if (l16 == 0)
#pragma unroll
    for (int r = 0; r < 8; ++r) {
      ssum[wave * 16 + r + 8 * half] = psum[r];
      ssq[wave * 16 + r + 8 * half]  = psq[r];
    }
  __syncthreads();

  float mu[8], rstd[8];
#pragma unroll
  for (int r = 0; r < 8; ++r) {
    const int rr = r + 8 * half;
    float s = 0.f, q = 0.f;
#pragma unroll
    for (int wv = 0; wv < 8; ++wv) { s += ssum[wv * 16 + rr]; q += ssq[wv * 16 + rr]; }
    const float m = s * (1.0f / D_MODEL);
    const float var = q * (1.0f / D_MODEL) - m * m;
    mu[r] = m;
    rstd[r] = rsqrtf(var + LN_EPS);
  }

#pragma unroll
  for (int t = 0; t < 8; ++t) {
    const int col = n0 + t * 16 + l16;
    const float g = gamma[col], be = beta[col];
#pragma unroll
    for (int r = 0; r < 8; ++r) {
      const int row = m0 + r + 8 * half;
      out[(size_t)row * D_MODEL + col] = (acc[t][r] - mu[r]) * rstd[r] * g + be;
    }
  }
}

// ---------------------------------------------------------------------------
// Launcher
// ---------------------------------------------------------------------------
extern "C" void kernel_launch(void* const* d_in, const int* in_sizes, int n_in,
                              void* d_out, int out_size, void* d_ws, size_t ws_size,
                              hipStream_t stream) {
  (void)in_sizes; (void)n_in; (void)out_size; (void)ws_size;
  const float* query = (const float*)d_in[0];
  const float* key   = (const float*)d_in[1];
  const float* value = (const float*)d_in[2];
  const float* Wq    = (const float*)d_in[3];
  const float* bq    = (const float*)d_in[4];
  const float* Wk    = (const float*)d_in[5];
  const float* bk    = (const float*)d_in[6];
  const float* Wv    = (const float*)d_in[7];
  const float* bv    = (const float*)d_in[8];
  const float* Wo    = (const float*)d_in[9];
  const float* bo    = (const float*)d_in[10];
  const float* gamma = (const float*)d_in[11];
  const float* beta  = (const float*)d_in[12];

  float* out  = (float*)d_out;                       // [B,S,D] = 4096*1024
  float* attn = out + (size_t)ROWS * D_MODEL;        // [B,H,S,S]

  const size_t MB = (size_t)1 << 20;
  char* ws = (char*)d_ws;                            // 64 MB bf16 staging
  bf16* Qb   = (bf16*)(ws);                          //  8 MB [b,h,s,dk]
  bf16* Kb   = (bf16*)(ws +  8 * MB);                //  8 MB [b,h,s,dk]
  bf16* Vtb  = (bf16*)(ws + 16 * MB);                //  8 MB [b,h,dk,s]
  bf16* ctx  = (bf16*)(ws + 24 * MB);                //  8 MB [b,s,d]
  bf16* Xq   = (bf16*)(ws + 32 * MB);                //  8 MB
  bf16* Xk   = (bf16*)(ws + 40 * MB);                //  8 MB
  bf16* Xv   = (bf16*)(ws + 48 * MB);                //  8 MB
  bf16* Wqb  = (bf16*)(ws + 56 * MB);                //  2 MB
  bf16* Wkb  = (bf16*)(ws + 58 * MB);                //  2 MB
  bf16* Wvb  = (bf16*)(ws + 60 * MB);                //  2 MB
  bf16* Wob  = (bf16*)(ws + 62 * MB);                //  2 MB

  const int NX = ROWS * D_MODEL;       // 4194304
  const int NW = D_MODEL * D_MODEL;    // 1048576
  cvt_kernel<<<NX / 1024, 256, 0, stream>>>(query, Xq, NX);
  cvt_kernel<<<NX / 1024, 256, 0, stream>>>(key,   Xk, NX);
  cvt_kernel<<<NX / 1024, 256, 0, stream>>>(value, Xv, NX);
  cvt_kernel<<<NW / 1024, 256, 0, stream>>>(Wq, Wqb, NW);
  cvt_kernel<<<NW / 1024, 256, 0, stream>>>(Wk, Wkb, NW);
  cvt_kernel<<<NW / 1024, 256, 0, stream>>>(Wv, Wvb, NW);
  cvt_kernel<<<NW / 1024, 256, 0, stream>>>(Wo, Wob, NW);

  dim3 pgrid(8, 32), pblock(256);
  proj_kernel<<<pgrid, pblock, 0, stream>>>(Xq, Wqb, bq, Qb, 0);
  proj_kernel<<<pgrid, pblock, 0, stream>>>(Xk, Wkb, bk, Kb, 1);
  proj_kernel<<<pgrid, pblock, 0, stream>>>(Xv, Wvb, bv, Vtb, 2);

  const size_t smem2 = (size_t)16 * ATTN_STRIDE * 2   // bf16 attn tile
                     + (size_t)16 * D_K * 4           // fp32 ctx tile
                     + (size_t)8 * 16 * 4;            // reduction scratch
  attn_kernel<<<BATCH * NUM_HEADS * (SEQ / 16), 256, smem2, stream>>>(
      Qb, Kb, Vtb, attn, ctx);

  outproj_ln_kernel<<<ROWS / 16, 256, 0, stream>>>(
      ctx, Wob, bo, query, gamma, beta, out);
}
